// CrossAxialMultiAttention_223338300233
// MI455X (gfx1250) — matrix operand order repaired and checked by reading, not compiled, not run on hardware
//
#include <hip/hip_runtime.h>
#include <hip/hip_bf16.h>
#include <math.h>

typedef __attribute__((ext_vector_type(16))) _Float16 v16h;
typedef __attribute__((ext_vector_type(8)))  float    v8f;

// Problem constants
#define C_DIM 256
#define NH    8
#define DH    32
#define B_SZ  2
#define M_AX  32
#define T_LEN 512

// ---------------------------------------------------------------------------
// Fragment loaders: f32 row-major source -> f16 WMMA fragments.
// A-fragment (16x32, MxK): lane m<16: K = {k..k+7} u {k+16..k+23};
//                          lane m+16: K = {k+8..k+15} u {k+24..k+31}
// B-fragment (32x16, KxN) for C = A*W^T, loaded from W rows (row = out col):
//                          lane n<16: K = {k..k+7} u {k+16..k+23}; lane n+16: K = {k+8..k+15} u {k+24..k+31}
// ---------------------------------------------------------------------------
static __device__ __forceinline__ v16h frag_a_f32(const float* __restrict__ base,
                                                  int ld, int row, int kk, int lane) {
  const float* p = base + (long)row * ld + kk + ((lane >> 4) << 3);
  float4 f0 = *(const float4*)(p + 0);
  float4 f1 = *(const float4*)(p + 4);
  float4 f2 = *(const float4*)(p + 16);
  float4 f3 = *(const float4*)(p + 20);
  v16h h;
  h[0]=(_Float16)f0.x;  h[1]=(_Float16)f0.y;  h[2]=(_Float16)f0.z;  h[3]=(_Float16)f0.w;
  h[4]=(_Float16)f1.x;  h[5]=(_Float16)f1.y;  h[6]=(_Float16)f1.z;  h[7]=(_Float16)f1.w;
  h[8]=(_Float16)f2.x;  h[9]=(_Float16)f2.y;  h[10]=(_Float16)f2.z; h[11]=(_Float16)f2.w;
  h[12]=(_Float16)f3.x; h[13]=(_Float16)f3.y; h[14]=(_Float16)f3.z; h[15]=(_Float16)f3.w;
  return h;
}

static __device__ __forceinline__ v16h frag_b_f32(const float* __restrict__ base,
                                                  int ld, int row, int kk, int lane) {
  const float* p = base + (long)row * ld + kk + ((lane >> 4) << 3);
  float4 f0 = *(const float4*)(p + 0);
  float4 f1 = *(const float4*)(p + 4);
  float4 f2 = *(const float4*)(p + 16);
  float4 f3 = *(const float4*)(p + 20);
  v16h h;
  h[0]=(_Float16)f0.x;  h[1]=(_Float16)f0.y;  h[2]=(_Float16)f0.z;  h[3]=(_Float16)f0.w;
  h[4]=(_Float16)f1.x;  h[5]=(_Float16)f1.y;  h[6]=(_Float16)f1.z;  h[7]=(_Float16)f1.w;
  h[8]=(_Float16)f2.x;  h[9]=(_Float16)f2.y;  h[10]=(_Float16)f2.z; h[11]=(_Float16)f2.w;
  h[12]=(_Float16)f3.x; h[13]=(_Float16)f3.y; h[14]=(_Float16)f3.z; h[15]=(_Float16)f3.w;
  return h;
}

// ---------------------------------------------------------------------------
// Generic batched NT GEMM: C = alpha * (A @ W^T) + bias_scale * bias[col]
// A: [Mrows, K] row-major (lda), W: [Ncols, K] row-major (ldb), C: (ldc).
// Batch index z = (zb*nMid + zm)*nInner + zn with independent strides.
// One wave -> a 16 x (16*TN) strip of C: one A-fragment feeds TN WMMAs per
// K-step (TN-fold reuse of A-side traffic). K must be a multiple of 32;
// Ncols must be a multiple of 16*TN.
// ---------------------------------------------------------------------------
template <int TN>
__global__ void __launch_bounds__(256)
ca_gemm_nt(const float* __restrict__ Abase, const float* __restrict__ Bbase,
           float* __restrict__ Cbase,
           int Mrows, int Ncols, int K, int lda, int ldb, int ldc,
           const float* __restrict__ bias, float bias_scale, float alpha,
           int nInner, int nMid,
           long sAb, long sAm, long sAn,
           long sBb, long sBm, long sBn,
           long sCb, long sCm, long sCn) {
  int z  = blockIdx.z;
  int zn = z % nInner;
  int zm = (z / nInner) % nMid;
  int zb = z / (nInner * nMid);
  const float* A = Abase + zb * sAb + zm * sAm + zn * sAn;
  const float* B = Bbase + zb * sBb + zm * sBm + zn * sBn;
  float*       C = Cbase + zb * sCb + zm * sCm + zn * sCn;

  int groupsN   = (Ncols >> 4) / TN;
  int numWaves  = (Mrows >> 4) * groupsN;
  int wave      = blockIdx.x * (blockDim.x >> 5) + (threadIdx.x >> 5);
  if (wave >= numWaves) return;          // wave-uniform: EXEC stays all-ones
  int tm  = wave / groupsN;
  int gn  = wave % groupsN;

  int lane  = threadIdx.x & 31;
  int rhalf = lane & 15;
  int arow  = (tm << 4) + rhalf;

  v8f acc[TN];
#pragma unroll
  for (int j = 0; j < TN; ++j) acc[j] = (v8f){};

  for (int kk = 0; kk < K; kk += 32) {
    v16h af = frag_a_f32(A, lda, arow, kk, lane);
#pragma unroll
    for (int j = 0; j < TN; ++j) {
      v16h bf = frag_b_f32(B, ldb, ((gn * TN + j) << 4) + rhalf, kk, lane);
      acc[j] = __builtin_amdgcn_wmma_f32_16x16x32_f16(false, af, false, bf,
                                                      (short)0, acc[j], false, false);
    }
  }

  int row0 = (tm << 4) + ((lane < 16) ? 0 : 8);
#pragma unroll
  for (int j = 0; j < TN; ++j) {
    int   col = ((gn * TN + j) << 4) + rhalf;
    float bv  = bias ? (bias_scale * bias[col]) : 0.0f;
#pragma unroll
    for (int i = 0; i < 8; ++i) {
      C[(long)(row0 + i) * ldc + col] = alpha * acc[j][i] + bv;
    }
  }
}

// ---------------------------------------------------------------------------
// Reduce over axial M: out[b, t, c] = sum_m in[b, m, t, c]
// ---------------------------------------------------------------------------
__global__ void ca_sum_m(const float* __restrict__ in, float* __restrict__ out) {
  const long inner = (long)T_LEN * C_DIM;           // 131072
  long i = (long)blockIdx.x * blockDim.x + threadIdx.x;
  if (i >= (long)B_SZ * inner) return;
  long b = i / inner, r = i % inner;
  const float* p = in + b * (long)M_AX * inner + r;
  float s = 0.0f;
#pragma unroll
  for (int m = 0; m < M_AX; ++m) s += p[(long)m * inner];
  out[i] = s;
}

// ---------------------------------------------------------------------------
// Row softmax over S=512. One wave per (b,n,t) row, in-place on w.
// Rows with b==0 (row < NH*T_LEN) are also written to a_out ([N,T,S]).
// ---------------------------------------------------------------------------
__global__ void __launch_bounds__(256)
ca_softmax(float* __restrict__ w, float* __restrict__ a_out) {
  int row  = blockIdx.x * (blockDim.x >> 5) + (threadIdx.x >> 5);
  int lane = threadIdx.x & 31;
  if (row >= B_SZ * NH * T_LEN) return;
  float* p = w + (long)row * T_LEN;

  float vals[16];
  float mx = -3.0e38f;
#pragma unroll
  for (int i = 0; i < 16; ++i) {
    vals[i] = p[lane + i * 32];
    mx = fmaxf(mx, vals[i]);
  }
  for (int off = 16; off; off >>= 1) mx = fmaxf(mx, __shfl_xor(mx, off, 32));

  float sum = 0.0f;
#pragma unroll
  for (int i = 0; i < 16; ++i) {
    vals[i] = __expf(vals[i] - mx);
    sum += vals[i];
  }
  for (int off = 16; off; off >>= 1) sum += __shfl_xor(sum, off, 32);
  float inv = 1.0f / sum;

  bool copy_a = (row < NH * T_LEN);
  float* q = a_out + (long)row * T_LEN;
#pragma unroll
  for (int i = 0; i < 16; ++i) {
    float r = vals[i] * inv;
    p[lane + i * 32] = r;
    if (copy_a) q[lane + i * 32] = r;
  }
}

// ---------------------------------------------------------------------------
// Per-head transpose of v: vT[((b*32+m)*8+n)*32 + d][s] = v_flat[(b*32+m)*512+s][n*32+d]
// Makes att = w @ v_head an NT GEMM with contiguous K loads.
// ---------------------------------------------------------------------------
__global__ void ca_transpose_v(const float* __restrict__ v, float* __restrict__ vT) {
  long i = (long)blockIdx.x * blockDim.x + threadIdx.x;
  const long total = (long)B_SZ * M_AX * NH * DH * T_LEN;
  if (i >= total) return;
  int  s  = (int)(i % T_LEN);
  long t1 = i / T_LEN;
  int  d  = (int)(t1 % DH);  t1 /= DH;
  int  n  = (int)(t1 % NH);  t1 /= NH;
  long bm = t1;                                  // b*32+m
  vT[i] = v[(bm * T_LEN + s) * C_DIM + n * DH + d];
}

// ---------------------------------------------------------------------------
extern "C" void kernel_launch(void* const* d_in, const int* in_sizes, int n_in,
                              void* d_out, int out_size, void* d_ws, size_t ws_size,
                              hipStream_t stream) {
  (void)in_sizes; (void)n_in; (void)out_size; (void)ws_size;
  const float* x  = (const float*)d_in[0];
  const float* y  = (const float*)d_in[1];
  const float* Wq = (const float*)d_in[2];
  const float* bq = (const float*)d_in[3];
  const float* Wk = (const float*)d_in[4];
  const float* bk = (const float*)d_in[5];
  const float* Wv = (const float*)d_in[6];
  const float* bv = (const float*)d_in[7];
  const float* Wp = (const float*)d_in[8];
  const float* bp = (const float*)d_in[9];

  float* out   = (float*)d_out;                                   // [2,32,512,256]
  float* a_out = out + (long)B_SZ * M_AX * T_LEN * C_DIM;         // [8,512,512]

  // Workspace layout (floats)
  float* ws    = (float*)d_ws;
  float* xsum  = ws;                       // 262144  [2,512,256]
  float* ysum  = xsum  + 262144;           // 262144
  float* qt    = ysum  + 262144;           // 262144  [2,512,256]
  float* kt    = qt    + 262144;           // 262144
  float* vbuf  = kt    + 262144;           // 8388608 [2*32*512, 256]
  float* wsc   = vbuf  + 8388608;          // 4194304 [2,8,512,512] scores -> softmax
  float* vT    = wsc   + 4194304;          // 8388608 [2,32,8,32,512]
  float* att   = vT    + 8388608;          // 8388608 [2*32*512, 256]

  const int TPB = 256;   // 8 waves per block

  // 1) axial sums
  ca_sum_m<<<dim3((B_SZ * T_LEN * C_DIM) / TPB), TPB, 0, stream>>>(x, xsum);
  ca_sum_m<<<dim3((B_SZ * T_LEN * C_DIM) / TPB), TPB, 0, stream>>>(y, ysum);

  // 2) qt = xsum @ Wq^T + 32*bq   [1024,256]; 64*4=256 waves -> 32 blocks
  ca_gemm_nt<4><<<dim3(32, 1, 1), TPB, 0, stream>>>(
      xsum, Wq, qt, 1024, 256, 256, 256, 256, 256, bq, (float)M_AX, 1.0f,
      1, 1, 0, 0, 0, 0, 0, 0, 0, 0, 0);
  // 3) kt = ysum @ Wk^T + 32*bk
  ca_gemm_nt<4><<<dim3(32, 1, 1), TPB, 0, stream>>>(
      ysum, Wk, kt, 1024, 256, 256, 256, 256, 256, bk, (float)M_AX, 1.0f,
      1, 1, 0, 0, 0, 0, 0, 0, 0, 0, 0);

  // 4) v = y_flat @ Wv^T + bv   [32768,256]; 2048*4=8192 waves -> 1024 blocks
  ca_gemm_nt<4><<<dim3(1024, 1, 1), TPB, 0, stream>>>(
      y, Wv, vbuf, 32768, 256, 256, 256, 256, 256, bv, 1.0f, 1.0f,
      1, 1, 0, 0, 0, 0, 0, 0, 0, 0, 0);

  // 5) scores[b,n] = (1/32) * qt_head @ kt_head^T    z = b*8+n, K=32
  //    32*8=256 waves per z -> 32 blocks, z=16
  ca_gemm_nt<4><<<dim3(32, 1, 16), TPB, 0, stream>>>(
      qt, kt, wsc, 512, 512, 32, 256, 256, 512, nullptr, 0.0f, 1.0f / 32.0f,
      /*nInner=*/8, /*nMid=*/1,
      /*A*/ (long)512 * 256, 0, 32,
      /*B*/ (long)512 * 256, 0, 32,
      /*C*/ (long)8 * 512 * 512, 0, (long)512 * 512);

  // 6) softmax rows (and emit `a` for batch 0)
  ca_softmax<<<dim3((B_SZ * NH * T_LEN) / 8), TPB, 0, stream>>>(wsc, a_out);

  // 7) per-head transpose of v
  ca_transpose_v<<<dim3((unsigned)((long)B_SZ * M_AX * NH * DH * T_LEN / TPB)),
                   TPB, 0, stream>>>(vbuf, vT);

  // 8) att[b,m,n] = w[b,n] @ vT[b,m,n]^T   z = ((b*32+m)*8+n), K=512
  //    Ncols=32 -> TN=2: 32*1=32 waves per z -> 4 blocks, z=512
  ca_gemm_nt<2><<<dim3(4, 1, 512), TPB, 0, stream>>>(
      wsc, vT, att, 512, 32, 512, 512, 512, 256, nullptr, 0.0f, 1.0f,
      /*nInner=*/8, /*nMid=*/32,
      /*A*/ (long)8 * 512 * 512, 0, (long)512 * 512,
      /*B*/ (long)NH * M_AX * DH * T_LEN, (long)NH * DH * T_LEN, (long)DH * T_LEN,
      /*C*/ (long)M_AX * 512 * 256, (long)512 * 256, 32);

  // 9) out = att @ Wp^T + bp   [32768,256]
  ca_gemm_nt<4><<<dim3(1024, 1, 1), TPB, 0, stream>>>(
      att, Wp, out, 32768, 256, 256, 256, 256, 256, bp, 1.0f, 1.0f,
      1, 1, 0, 0, 0, 0, 0, 0, 0, 0, 0);
}
